// BertFFNTrainableModule_32023276159360
// MI455X (gfx1250) — matrix operand order repaired and checked by reading, not compiled, not run on hardware
//
#include <hip/hip_runtime.h>
#include <hip/hip_bf16.h>

typedef __attribute__((ext_vector_type(16))) __bf16    v16bf;
typedef __attribute__((ext_vector_type(8)))  float     v8f;
typedef __attribute__((ext_vector_type(8)))  unsigned  v8u;

// ---------------- workspace layout (float units) ----------------
// Wd'  [768][16]  g1-scaled W_down
// dbias[16]       b1@W_down + b_down
// key' [64][16]   g2-scaled (memory@W_k+b_k), rows 50..63 = 0
// sbias[64]       b2 @ key^T, rows 50..63 = 0
// val  [64][16]   memory@W_v+b_v, rows 50..63 = 0
// Wu'  [16][768]  g3-scaled W_up
// obias[768]      b3@W_up + b_up
#define OFF_WD   0
#define OFF_DB   12288
#define OFF_KEY  12304
#define OFF_SB   13328
#define OFF_VAL  13392
#define OFF_WU   14416
#define OFF_OB   26704

__global__ void prep_kernel(const float* __restrict__ g1, const float* __restrict__ b1,
                            const float* __restrict__ Wd, const float* __restrict__ bd,
                            const float* __restrict__ g2, const float* __restrict__ b2,
                            const float* __restrict__ mem,
                            const float* __restrict__ Wk, const float* __restrict__ bk,
                            const float* __restrict__ Wv, const float* __restrict__ bv,
                            const float* __restrict__ g3, const float* __restrict__ b3,
                            const float* __restrict__ Wu, const float* __restrict__ bu,
                            float* __restrict__ ws) {
  const int tid = threadIdx.x + blockIdx.x * blockDim.x;
  const int nth = blockDim.x * gridDim.x;

  for (int i = tid; i < 12288; i += nth) {
    ws[OFF_WD + i] = g1[i >> 4] * Wd[i];          // [768][16]
    ws[OFF_WU + i] = g3[i / 768] * Wu[i];         // [16][768]
  }
  for (int n = tid; n < 16; n += nth) {
    float s = bd[n];
    for (int k = 0; k < 768; ++k) s += b1[k] * Wd[k * 16 + n];
    ws[OFF_DB + n] = s;
  }
  for (int i = tid; i < 1024; i += nth) {
    const int m = i >> 4, d = i & 15;
    float kv = 0.f, vv = 0.f;
    if (m < 50) {
      kv = bk[d]; vv = bv[d];
      for (int e = 0; e < 16; ++e) {
        const float me = mem[m * 16 + e];
        kv += me * Wk[e * 16 + d];
        vv += me * Wv[e * 16 + d];
      }
      kv *= g2[d];
    }
    ws[OFF_KEY + i] = kv;
    ws[OFF_VAL + i] = vv;
  }
  for (int m = tid; m < 64; m += nth) {
    float s = 0.f;
    if (m < 50) {
      for (int d = 0; d < 16; ++d) {
        float kv = bk[d];
        for (int e = 0; e < 16; ++e) kv += mem[m * 16 + e] * Wk[e * 16 + d];
        s += b2[d] * kv;
      }
    }
    ws[OFF_SB + m] = s;
  }
  for (int c = tid; c < 768; c += nth) {
    float s = bu[c];
    for (int k = 0; k < 16; ++k) s += b3[k] * Wu[k * 768 + c];
    ws[OFF_OB + c] = s;
  }
}

// ---------------- helpers ----------------
__device__ __forceinline__ unsigned pack2(float a, float b) {
  const unsigned short ua = __builtin_bit_cast(unsigned short, (__bf16)a);
  const unsigned short ub = __builtin_bit_cast(unsigned short, (__bf16)b);
  return (unsigned)ua | ((unsigned)ub << 16);
}

__device__ __forceinline__ v16bf zero_frag() {
  v16bf v;
#pragma unroll
  for (int i = 0; i < 16; ++i) v[i] = (__bf16)0.0f;
  return v;
}

// Load 8 pre-swizzled LDS words (16 bf16) as a B fragment.
__device__ __forceinline__ v16bf load_bfrag(const unsigned* p) {
  const uint4 a = *(const uint4*)p;
  const uint4 b = *(const uint4*)(p + 4);
  v8u w; w[0]=a.x; w[1]=a.y; w[2]=a.z; w[3]=a.w; w[4]=b.x; w[5]=b.y; w[6]=b.z; w[7]=b.w;
  return __builtin_bit_cast(v16bf, w);
}

__device__ __forceinline__ void cvt8(v16bf& v, int base, float4 a, float4 b, float mu, float rs) {
  v[base + 0] = (__bf16)((a.x - mu) * rs);
  v[base + 1] = (__bf16)((a.y - mu) * rs);
  v[base + 2] = (__bf16)((a.z - mu) * rs);
  v[base + 3] = (__bf16)((a.w - mu) * rs);
  v[base + 4] = (__bf16)((b.x - mu) * rs);
  v[base + 5] = (__bf16)((b.y - mu) * rs);
  v[base + 6] = (__bf16)((b.z - mu) * rs);
  v[base + 7] = (__bf16)((b.w - mu) * rs);
}

#define WMMA_BF16(A, B, C) \
  __builtin_amdgcn_wmma_f32_16x16x32_bf16(false, (A), false, (B), (short)0, (C), false, false)

// ---------------- fused main kernel ----------------
// grid = 256 blocks x 256 threads; 8 waves/block, 16 rows/wave, 128 rows/block.
__global__ void __launch_bounds__(256)
ffn_mem_kernel(const float* __restrict__ hidden, const float* __restrict__ ws,
               float* __restrict__ out) {
  extern __shared__ unsigned char smem_raw[];
  unsigned* sWd  = (unsigned*)smem_raw;   // [24][32][8] words (down B frags)
  unsigned* sWu  = sWd  + 6144;           // [48][16][8] words (up   B frags, K 0..15)
  unsigned* sKey = sWu  + 6144;           // [ 4][16][8] words (score B frags, K 0..15)
  unsigned* sVal = sKey + 512;            // [ 2][32][8] words (attnV B frags)
  float* sDb = (float*)(sVal + 512);      // 16
  float* sSb = sDb + 16;                  // 64
  float* sOb = sSb + 64;                  // 768
  float* sScore = sOb + 768;              // 8 waves * 16 * 64
  float* sQ     = sScore + 8 * 1024;      // 8 waves * 16 * 16

  const int tid = threadIdx.x;

  // ---- stage weights into pre-swizzled B-fragment word order ----
  const float* wsWd = ws + OFF_WD;
  const float* wsWu = ws + OFF_WU;
  const float* wsKey = ws + OFF_KEY;
  const float* wsVal = ws + OFF_VAL;
  for (int w = tid; w < 6144; w += 256) {   // down: word(c,L,j) = Wd'[k..k+1][n]
    const int c = w >> 8, L = (w >> 3) & 31, j = w & 7;
    const int k0 = c * 32 + ((L & 16) ? 8 : 0) + 2 * j + ((j & 4) ? 8 : 0), n = L & 15;
    sWd[w] = pack2(wsWd[k0 * 16 + n], wsWd[(k0 + 1) * 16 + n]);
  }
  for (int w = tid; w < 6144; w += 256) {   // up: word(t,n,j) = Wu'[2j..2j+1][t*16+n]
    const int t = w >> 7, n = (w >> 3) & 15, j = w & 7;
    const int col = t * 16 + n;
    sWu[w] = pack2(wsWu[(2 * j) * 768 + col], wsWu[(2 * j + 1) * 768 + col]);
  }
  for (int w = tid; w < 512; w += 256) {    // key^T: word(t,n,j) = key'[t*16+n][2j..2j+1]
    const int t = w >> 7, n = (w >> 3) & 15, j = w & 7;
    const int m = t * 16 + n;
    sKey[w] = pack2(wsKey[m * 16 + 2 * j], wsKey[m * 16 + 2 * j + 1]);
  }
  for (int w = tid; w < 512; w += 256) {    // val: word(c,L,j) = val[m..m+1][n]
    const int c = w >> 8, L = (w >> 3) & 31, j = w & 7;
    const int m0 = c * 32 + ((L & 16) ? 8 : 0) + 2 * j + ((j & 4) ? 8 : 0), n = L & 15;
    sVal[w] = pack2(wsVal[m0 * 16 + n], wsVal[(m0 + 1) * 16 + n]);
  }
  for (int i = tid; i < 16;  i += 256) sDb[i] = ws[OFF_DB + i];
  for (int i = tid; i < 64;  i += 256) sSb[i] = ws[OFF_SB + i];
  for (int i = tid; i < 768; i += 256) sOb[i] = ws[OFF_OB + i];
  __syncthreads();

  const int wave = tid >> 5, lane = tid & 31;
  const int rl = lane & 15;          // row in tile (A) / col (B,C)
  const int hi = lane >> 4;          // lane half
  const int rowBase = blockIdx.x * 128 + wave * 16;
  float* myScore = sScore + wave * 1024;
  float* myQ     = sQ + wave * 256;

  // ---- ln1 stats over H=768 (2 lanes per row) ----
  float s = 0.f, ss = 0.f;
  {
    const float* hp = hidden + (size_t)(rowBase + rl) * 768 + hi * 384;
#pragma unroll 4
    for (int i = 0; i < 384; i += 4) {
      const float4 v = *(const float4*)(hp + i);
      s  += v.x + v.y + v.z + v.w;
      ss += v.x * v.x + v.y * v.y + v.z * v.z + v.w * v.w;
    }
  }
  s  += __shfl_xor(s, 16, 32);
  ss += __shfl_xor(ss, 16, 32);
  const float mean = s * (1.f / 768.f);
  const float rsig = rsqrtf(ss * (1.f / 768.f) - mean * mean + 1e-12f);

  // ---- down GEMM: d = ln1(h) @ Wd' + dbias   (24 x wmma bf16 K=32) ----
  v8f accd;
  {
    const float db = sDb[rl];
#pragma unroll
    for (int r = 0; r < 8; ++r) accd[r] = db;
  }
  {
    const float* arow = hidden + (size_t)(rowBase + rl) * 768 + (hi ? 8 : 0);
#pragma unroll 2
    for (int c = 0; c < 24; ++c) {
      const float* p = arow + c * 32;
      const float4 x0 = *(const float4*)(p);
      const float4 x1 = *(const float4*)(p + 4);
      const float4 x2 = *(const float4*)(p + 16);
      const float4 x3 = *(const float4*)(p + 20);
      v16bf va;
      cvt8(va, 0, x0, x1, mean, rsig);
      cvt8(va, 8, x2, x3, mean, rsig);
      const v16bf vb = load_bfrag(sWd + c * 256 + lane * 8);
      accd = WMMA_BF16(va, vb, accd);
    }
  }

  // ---- ln2 on d (through LDS C->A relayout) ----
#pragma unroll
  for (int r = 0; r < 8; ++r) myQ[(r + hi * 8) * 16 + rl] = accd[r];
  __syncthreads();
  float qs = 0.f, qss = 0.f;
#pragma unroll
  for (int i = 0; i < 16; i += 4) {
    const float4 v = *(const float4*)(myQ + rl * 16 + i);
    qs  += v.x + v.y + v.z + v.w;
    qss += v.x * v.x + v.y * v.y + v.z * v.z + v.w * v.w;
  }
  const float qm = qs * (1.f / 16.f);
  const float qr = rsqrtf(qss * (1.f / 16.f) - qm * qm + 1e-12f);
  v16bf vq = zero_frag();                 // K 16..31 padded zero
  if (lane < 16) {
    const float* p = myQ + rl * 16;
    cvt8(vq, 0, *(const float4*)p, *(const float4*)(p + 4), qm, qr);
    cvt8(vq, 8, *(const float4*)(p + 8), *(const float4*)(p + 12), qm, qr);
  }

  // ---- scores = q @ key'^T + sbias   (4 N-tiles, K=16 padded) ----
  v8f accs[4];
#pragma unroll
  for (int t = 0; t < 4; ++t) {
    const float sb = sSb[t * 16 + rl];
#pragma unroll
    for (int r = 0; r < 8; ++r) accs[t][r] = sb;
    const v16bf vb = (lane < 16) ? load_bfrag(sKey + t * 128 + rl * 8) : zero_frag();
    accs[t] = WMMA_BF16(vq, vb, accs[t]);
  }
#pragma unroll
  for (int t = 0; t < 4; ++t)
#pragma unroll
    for (int r = 0; r < 8; ++r)
      myScore[(r + hi * 8) * 64 + t * 16 + rl] = accs[t][r];
  __syncthreads();

  // ---- softmax over M=50 (one lane per row) ----
  if (lane < 16) {
    float* row = myScore + rl * 64;
    float mx = -1e30f;
    for (int m = 0; m < 50; ++m) mx = fmaxf(mx, row[m]);
    float sum = 0.f;
    for (int m = 0; m < 50; ++m) { const float e = __expf(row[m] - mx); row[m] = e; sum += e; }
    const float inv = 1.f / sum;
    for (int m = 0; m < 50; ++m) row[m] *= inv;
    for (int m = 50; m < 64; ++m) row[m] = 0.f;
  }
  __syncthreads();

  // ---- mem_out = probs @ val   (K=64 -> 2 x wmma) ----
  v8f accv;
#pragma unroll
  for (int r = 0; r < 8; ++r) accv[r] = 0.f;
#pragma unroll
  for (int c = 0; c < 2; ++c) {
    const float* p = myScore + rl * 64 + c * 32 + (hi ? 8 : 0);
    v16bf va;
    cvt8(va, 0, *(const float4*)(p),      *(const float4*)(p + 4),  0.f, 1.f);
    cvt8(va, 8, *(const float4*)(p + 16), *(const float4*)(p + 20), 0.f, 1.f);
    const v16bf vb = load_bfrag(sVal + c * 256 + lane * 8);
    accv = WMMA_BF16(va, vb, accv);
  }

  // ---- ln3 on mem_out ----
#pragma unroll
  for (int r = 0; r < 8; ++r) myQ[(r + hi * 8) * 16 + rl] = accv[r];
  __syncthreads();
  float zs = 0.f, zss = 0.f;
#pragma unroll
  for (int i = 0; i < 16; i += 4) {
    const float4 v = *(const float4*)(myQ + rl * 16 + i);
    zs  += v.x + v.y + v.z + v.w;
    zss += v.x * v.x + v.y * v.y + v.z * v.z + v.w * v.w;
  }
  const float zm = zs * (1.f / 16.f);
  const float zr = rsqrtf(zss * (1.f / 16.f) - zm * zm + 1e-12f);
  v16bf vz = zero_frag();
  if (lane < 16) {
    const float* p = myQ + rl * 16;
    cvt8(vz, 0, *(const float4*)p, *(const float4*)(p + 4), zm, zr);
    cvt8(vz, 8, *(const float4*)(p + 8), *(const float4*)(p + 12), zm, zr);
  }

  // ---- up GEMM: out = ln3 @ Wu' + obias  (48 N-tiles, K=16 padded) ----
#pragma unroll 2
  for (int t = 0; t < 48; ++t) {
    const v16bf vb = (lane < 16) ? load_bfrag(sWu + t * 128 + rl * 8) : zero_frag();
    v8f acco;
    const float ob = sOb[t * 16 + rl];
#pragma unroll
    for (int r = 0; r < 8; ++r) acco[r] = ob;
    acco = WMMA_BF16(vz, vb, acco);
#pragma unroll
    for (int r = 0; r < 8; ++r)
      out[(size_t)(rowBase + r + hi * 8) * 768 + t * 16 + rl] = acco[r];
  }
}

extern "C" void kernel_launch(void* const* d_in, const int* in_sizes, int n_in,
                              void* d_out, int out_size, void* d_ws, size_t ws_size,
                              hipStream_t stream) {
  const float* hidden = (const float*)d_in[0];
  const float* g1 = (const float*)d_in[1];
  const float* b1 = (const float*)d_in[2];
  const float* Wd = (const float*)d_in[3];
  const float* bd = (const float*)d_in[4];
  const float* g2 = (const float*)d_in[5];
  const float* b2 = (const float*)d_in[6];
  const float* mem = (const float*)d_in[7];
  const float* Wk = (const float*)d_in[8];
  const float* bk = (const float*)d_in[9];
  const float* Wv = (const float*)d_in[10];
  const float* bv = (const float*)d_in[11];
  const float* g3 = (const float*)d_in[12];
  const float* b3 = (const float*)d_in[13];
  const float* Wu = (const float*)d_in[14];
  const float* bu = (const float*)d_in[15];
  float* ws = (float*)d_ws;
  float* out = (float*)d_out;

  prep_kernel<<<4, 256, 0, stream>>>(g1, b1, Wd, bd, g2, b2, mem, Wk, bk, Wv, bv,
                                     g3, b3, Wu, bu, ws);

  const int smem = 97600;  // ~95.3 KB dynamic LDS (WGP allows up to 320 KB)
  hipFuncSetAttribute((const void*)ffn_mem_kernel,
                      hipFuncAttributeMaxDynamicSharedMemorySize, smem);
  ffn_mem_kernel<<<256, 256, smem, stream>>>(hidden, ws, out);
}
